// DPRIndexModule_35596688949677
// MI455X (gfx1250) — compile-verified, matrix operand order checked by reading, not run on hardware
//
#include <hip/hip_runtime.h>
#include <hip/hip_bf16.h>

#define B_Q    64          // queries
#define D_DIM  768         // embedding dim
#define TILE_N 128         // docs per block
#define KC     32          // K chunk (one bf16 WMMA K-step)
#define NKC    (D_DIM / KC)
#define KDOC   10

typedef __attribute__((ext_vector_type(16))) __bf16 v16bf;
typedef __attribute__((ext_vector_type(8)))  __bf16 v8bf;
typedef __attribute__((ext_vector_type(8)))  float  v8f;

// ---------------------------------------------------------------------------
// CDNA5 async global->LDS copy (ASYNCcnt path). LDS byte address is the low
// 32 bits of the generic pointer.
// ---------------------------------------------------------------------------
__device__ __forceinline__ void dpr_async_copy16(void* lds_dst, const void* gsrc) {
    unsigned lds_off = (unsigned)(uintptr_t)lds_dst;
    asm volatile("global_load_async_to_lds_b128 %0, %1, off"
                 :: "v"(lds_off), "v"(gsrc) : "memory");
}
__device__ __forceinline__ void dpr_wait_async() {
    asm volatile("s_wait_asynccnt 0" ::: "memory");
}

// ---------------------------------------------------------------------------
// Kernel 1: L2-normalize the 64 query rows and convert to bf16.
// ---------------------------------------------------------------------------
__global__ void __launch_bounds__(256)
dpr_norm_q(const float* __restrict__ q, __bf16* __restrict__ qbf) {
    const int b = blockIdx.x;
    const int t = threadIdx.x;
    const float* row = q + (size_t)b * D_DIM;

    float v0 = row[t], v1 = row[t + 256], v2 = row[t + 512];
    float s = v0 * v0 + v1 * v1 + v2 * v2;
    #pragma unroll
    for (int o = 16; o > 0; o >>= 1) s += __shfl_down(s, o, 32);

    __shared__ float red[8];
    __shared__ float rnorm;
    if ((t & 31) == 0) red[t >> 5] = s;
    __syncthreads();
    if (t == 0) {
        float x = 0.f;
        #pragma unroll
        for (int i = 0; i < 8; ++i) x += red[i];
        rnorm = rsqrtf(x + 1e-30f);
    }
    __syncthreads();
    const float rn = rnorm;
    qbf[(size_t)b * D_DIM + t      ] = (__bf16)(v0 * rn);
    qbf[(size_t)b * D_DIM + t + 256] = (__bf16)(v1 * rn);
    qbf[(size_t)b * D_DIM + t + 512] = (__bf16)(v2 * rn);
}

// ---------------------------------------------------------------------------
// Fragment loader: 16-bit A-matrix 16x32 layout (B column-major mirrors it).
// Lane 0-15 -> K {0..7,16..23}; lane 16-31 -> +8. base = [16 rows][KC] bf16.
// ---------------------------------------------------------------------------
__device__ __forceinline__ v16bf dpr_load_frag(const __bf16* base, int lane) {
    const int r  = lane & 15;
    const int ko = (lane & 16) ? 8 : 0;
    const __bf16* p = base + r * KC + ko;
    v8bf lo = *(const v8bf*)(p);        // K = ko .. ko+7
    v8bf hi = *(const v8bf*)(p + 16);   // K = ko+16 .. ko+23
    v16bf out;
    #pragma unroll
    for (int i = 0; i < 8; ++i) { out[i] = lo[i]; out[i + 8] = hi[i]; }
    return out;
}

__device__ __forceinline__ void dpr_cvt_store(__bf16* d, float4 r0, float4 r1,
                                              float4 r2, float4 r3, float& sq) {
    float4 rr[4] = {r0, r1, r2, r3};
    #pragma unroll
    for (int u = 0; u < 4; ++u) {
        float4 f = rr[u];
        sq += f.x * f.x + f.y * f.y + f.z * f.z + f.w * f.w;
        d[u * 4 + 0] = (__bf16)f.x; d[u * 4 + 1] = (__bf16)f.y;
        d[u * 4 + 2] = (__bf16)f.z; d[u * 4 + 3] = (__bf16)f.w;
    }
}

__device__ __forceinline__ void dpr_topk_insert(float v, int idx,
                                                float (&tv)[KDOC], int (&ti)[KDOC]) {
    if (v <= tv[KDOC - 1]) return;
    int j = KDOC - 1;
    while (j > 0 && tv[j - 1] < v) { tv[j] = tv[j - 1]; ti[j] = ti[j - 1]; --j; }
    tv[j] = v; ti[j] = idx;
}

// ---------------------------------------------------------------------------
// Kernel 2: stream a 128-doc tile (fp32 -> bf16, double-buffered LDS),
// accumulate q·d with bf16 WMMA over K=768 overlapping next-chunk loads with
// current-chunk WMMA, scale by rsqrt(|d|^2), block-local top-10 per query.
// ---------------------------------------------------------------------------
__global__ void __launch_bounds__(256)
dpr_sim_topk(const __bf16* __restrict__ qbf, const float* __restrict__ docs,
             int N, float* __restrict__ pvals, int* __restrict__ pidx) {
    __shared__ __bf16 sQ[2][B_Q * KC];     // 8 KB
    __shared__ __bf16 sD[2][TILE_N * KC];  // 16 KB
    __shared__ float  sNorm2[TILE_N];      // 0.5 KB
    __shared__ float  sSims[B_Q * TILE_N]; // 32 KB

    const int t    = threadIdx.x;
    const int blk  = blockIdx.x;
    const int n0   = blk * TILE_N;
    const int wave = t >> 5;
    const int lane = t & 31;
    const int m_tile  = wave & 3;   // which 16 queries
    const int n_group = wave >> 2;  // which group of 4 doc sub-tiles

    // doc staging: 2 threads per doc row, 16 contiguous K each.
    // OOB rows are clamped to row N-1 (always-legal load); their sims are
    // filtered by the gd < N check in the top-k scan.
    const int  dloc = t >> 1;
    const int  koff = (t & 1) * 16;
    long gdoc = (long)n0 + dloc;
    if (gdoc >= (long)N) gdoc = (long)N - 1;
    const float* dsrc = docs + gdoc * (long)D_DIM;

    // query staging: 4 threads per query row, 16 B (8 bf16) each, async copy
    const int qrow = t >> 2;
    const int qk   = (t & 3) * 8;
    const __bf16* qsrc = qbf + (size_t)qrow * D_DIM + qk;

    v8f acc[4] = {};
    float sq = 0.0f;

    // ---- prologue: stage chunk 0 into buffer 0
    {
        const float4* p = (const float4*)(dsrc + koff);
        float4 r0 = p[0], r1 = p[1], r2 = p[2], r3 = p[3];
        dpr_async_copy16(&sQ[0][qrow * KC + qk], qsrc);
        dpr_cvt_store(&sD[0][dloc * KC + koff], r0, r1, r2, r3, sq);
        dpr_wait_async();
    }
    if (t < TILE_N) sNorm2[t] = 0.0f;
    __syncthreads();

    for (int kc = 0; kc < NKC; ++kc) {
        const int  cur  = kc & 1;
        const bool more = (kc + 1 < NKC);
        float4 r0, r1, r2, r3;
        if (more) {
            // issue next chunk's global loads + async q copy before the WMMAs
            const float4* p = (const float4*)(dsrc + (kc + 1) * KC + koff);
            r0 = p[0]; r1 = p[1]; r2 = p[2]; r3 = p[3];
            dpr_async_copy16(&sQ[cur ^ 1][qrow * KC + qk], qsrc + (size_t)(kc + 1) * KC);
            if (kc + 2 < NKC) __builtin_prefetch(dsrc + (kc + 2) * KC + koff, 0, 1);
        }

        v16bf a = dpr_load_frag(&sQ[cur][m_tile * 16 * KC], lane);
        #pragma unroll
        for (int j = 0; j < 4; ++j) {
            v16bf b = dpr_load_frag(&sD[cur][(n_group * 4 + j) * 16 * KC], lane);
            acc[j] = __builtin_amdgcn_wmma_f32_16x16x32_bf16(
                false, a, false, b, (short)0, acc[j], false, false);
        }

        if (more) {
            dpr_cvt_store(&sD[cur ^ 1][dloc * KC + koff], r0, r1, r2, r3, sq);
            dpr_wait_async();
        }
        __syncthreads();
    }

    atomicAdd(&sNorm2[dloc], sq);          // ds_add_f32
    __syncthreads();

    // scale by rsqrt(|d|^2) and scatter C tiles to LDS.
    // C layout: VGPR v, lanes 0-15 -> M=v, lanes 16-31 -> M=v+8; N = lane&15
    const int col = lane & 15;
    const int rh  = (lane & 16) ? 8 : 0;
    #pragma unroll
    for (int j = 0; j < 4; ++j) {
        const int nl = (n_group * 4 + j) * 16 + col;
        const float rn = rsqrtf(sNorm2[nl] + 1e-30f);
        #pragma unroll
        for (int v = 0; v < 8; ++v)
            sSims[(m_tile * 16 + rh + v) * TILE_N + nl] = acc[j][v] * rn;
    }
    __syncthreads();

    // block-local top-10 per query (threads 0..63, one query each)
    if (t < B_Q) {
        float tv[KDOC]; int ti[KDOC];
        #pragma unroll
        for (int j = 0; j < KDOC; ++j) { tv[j] = -INFINITY; ti[j] = -1; }
        for (int n = 0; n < TILE_N; ++n) {
            const int gd = n0 + n;
            if (gd < N) dpr_topk_insert(sSims[t * TILE_N + n], gd, tv, ti);
        }
        const size_t base = (size_t)blk * (B_Q * KDOC) + (size_t)t * KDOC;
        #pragma unroll
        for (int j = 0; j < KDOC; ++j) { pvals[base + j] = tv[j]; pidx[base + j] = ti[j]; }
    }
}

// ---------------------------------------------------------------------------
// Kernel 3: one block per query; reduce NB*10 candidates to the final top-10.
// ---------------------------------------------------------------------------
__global__ void __launch_bounds__(256)
dpr_final_topk(const float* __restrict__ pvals, const int* __restrict__ pidx,
               int NB, float* __restrict__ out) {
    const int q = blockIdx.x;
    const int t = threadIdx.x;
    const int C = NB * KDOC;

    float tv[KDOC]; int ti[KDOC];
    #pragma unroll
    for (int j = 0; j < KDOC; ++j) { tv[j] = -INFINITY; ti[j] = -1; }

    for (int i = t; i < C; i += 256) {
        const int blk = i / KDOC;
        const int j   = i - blk * KDOC;
        const size_t off = (size_t)blk * (B_Q * KDOC) + (size_t)q * KDOC + j;
        dpr_topk_insert(pvals[off], pidx[off], tv, ti);
    }

    __shared__ float lv[256 * KDOC];
    __shared__ int   li[256 * KDOC];
    #pragma unroll
    for (int j = 0; j < KDOC; ++j) { lv[t * KDOC + j] = tv[j]; li[t * KDOC + j] = ti[j]; }
    __syncthreads();

    if (t == 0) {
        float fv[KDOC]; int fi[KDOC];
        #pragma unroll
        for (int j = 0; j < KDOC; ++j) { fv[j] = -INFINITY; fi[j] = -1; }
        for (int i = 0; i < 256 * KDOC; ++i) dpr_topk_insert(lv[i], li[i], fv, fi);
        #pragma unroll
        for (int j = 0; j < KDOC; ++j) {
            out[q * KDOC + j]              = fv[j];          // top_vals
            out[B_Q * KDOC + q * KDOC + j] = (float)fi[j];   // top_idx (exact < 2^24)
        }
    }
}

// ---------------------------------------------------------------------------
extern "C" void kernel_launch(void* const* d_in, const int* in_sizes, int n_in,
                              void* d_out, int out_size, void* d_ws, size_t ws_size,
                              hipStream_t stream) {
    const float* q    = (const float*)d_in[0];
    const float* docs = (const float*)d_in[1];
    const int N  = in_sizes[1] / D_DIM;
    const int NB = (N + TILE_N - 1) / TILE_N;

    char* ws = (char*)d_ws;
    __bf16* qbf   = (__bf16*)ws;                             // 96 KB
    float*  pvals = (float*)(ws + (size_t)128 * 1024);       // NB*640 floats
    int*    pidx  = (int*)(ws + (size_t)128 * 1024 + (size_t)NB * B_Q * KDOC * sizeof(float));

    dpr_norm_q<<<B_Q, 256, 0, stream>>>(q, qbf);
    dpr_sim_topk<<<NB, 256, 0, stream>>>(qbf, docs, N, pvals, pidx);
    dpr_final_topk<<<B_Q, 256, 0, stream>>>(pvals, pidx, NB, (float*)d_out);
}
